// XTransformerMaskRes_23201413333432
// MI455X (gfx1250) — matrix operand order repaired and checked by reading, not compiled, not run on hardware
//
#include <hip/hip_runtime.h>
#include <hip/hip_bf16.h>

typedef _Float16 f16;
typedef __attribute__((ext_vector_type(16))) _Float16 v16h;
typedef __attribute__((ext_vector_type(4)))  _Float16 v4h;
typedef __attribute__((ext_vector_type(8)))  float    v8f;

#define S_  512
#define B_  64
#define D_  128
#define H_  8
#define FF_ 1024
#define L_  6

// ---------------------------------------------------------------------------
// WMMA fragment loaders (CDNA5 16x16x32 f16, wave32; A/B k order contiguous, not ISA table order)
// A 16x32: lane L holds row m=L&15; vgpr v holds K = 16*(L>=16) + 2*v {+0,+1}
// B 32x16: lane L holds col n=L&15; vgpr v holds K = 16*(L>=16) + 2*v {+0,+1}
// C/D 16x16 f32: lane L holds col n=L&15; vgpr i holds row m = i + 8*(L>=16)
// ---------------------------------------------------------------------------
__device__ __forceinline__ v16h frag_a(const f16* A, int lda) {
  int lane = threadIdx.x & 31;
  const f16* row = A + (size_t)(lane & 15) * lda;
  int hh = (lane >> 4) << 4;
  v16h a;
#pragma unroll
  for (int v = 0; v < 8; ++v) {
    int kb = hh + (v << 1);
    a[2*v]   = row[kb];
    a[2*v+1] = row[kb+1];
  }
  return a;
}

__device__ __forceinline__ v16h frag_a_pad(const f16* A, int lda, int kmax) {
  int lane = threadIdx.x & 31;
  const f16* row = A + (size_t)(lane & 15) * lda;
  int hh = (lane >> 4) << 4;
  v16h a;
#pragma unroll
  for (int v = 0; v < 8; ++v) {
    int kb = hh + (v << 1);
    a[2*v]   = (kb     < kmax) ? row[kb]   : (f16)0;
    a[2*v+1] = (kb + 1 < kmax) ? row[kb+1] : (f16)0;
  }
  return a;
}

// B = W^T where W is stored N x K row-major (PyTorch-style weight)
__device__ __forceinline__ v16h frag_bT(const f16* W, int ldw) {
  int lane = threadIdx.x & 31;
  const f16* row = W + (size_t)(lane & 15) * ldw;
  int hh = (lane >> 4) << 4;
  v16h b;
#pragma unroll
  for (int v = 0; v < 8; ++v) {
    int kb = hh + (v << 1);
    b[2*v]   = row[kb];
    b[2*v+1] = row[kb+1];
  }
  return b;
}

__device__ __forceinline__ v16h frag_bT_pad(const f16* W, int ldw, int kmax) {
  int lane = threadIdx.x & 31;
  const f16* row = W + (size_t)(lane & 15) * ldw;
  int hh = (lane >> 4) << 4;
  v16h b;
#pragma unroll
  for (int v = 0; v < 8; ++v) {
    int kb = hh + (v << 1);
    b[2*v]   = (kb     < kmax) ? row[kb]   : (f16)0;
    b[2*v+1] = (kb + 1 < kmax) ? row[kb+1] : (f16)0;
  }
  return b;
}

// B stored directly K x N row-major
__device__ __forceinline__ v16h frag_b_dir(const f16* Bm, int ldb) {
  int lane = threadIdx.x & 31;
  int n = lane & 15;
  int hh = (lane >> 4) << 4;
  v16h b;
#pragma unroll
  for (int v = 0; v < 8; ++v) {
    int kb = hh + (v << 1);
    b[2*v]   = Bm[(size_t)kb * ldb + n];
    b[2*v+1] = Bm[(size_t)(kb + 1) * ldb + n];
  }
  return b;
}

__device__ __forceinline__ v8f wmma32(v16h a, v16h b, v8f c) {
  return __builtin_amdgcn_wmma_f32_16x16x32_f16(false, a, false, b, (short)0, c,
                                                false, false);
}

// ---------------------------------------------------------------------------
// Utility kernels
// ---------------------------------------------------------------------------
__global__ __launch_bounds__(256) void k_cvt(const float* __restrict__ s,
                                             f16* __restrict__ d, int n) {
  int i = blockIdx.x * 256 + threadIdx.x;
  if (i < n) d[i] = (f16)s[i];
}

// src (S,B,D) f32 -> X (B,S,D) f32 + f16
__global__ __launch_bounds__(256) void k_transpose(const float* __restrict__ src,
                                                   float* __restrict__ X,
                                                   f16* __restrict__ Xh) {
  int i = blockIdx.x * 256 + threadIdx.x;          // output index over B*S*D
  int bb = i >> 16;                                // / (512*128)
  int s  = (i >> 7) & 511;
  int dd = i & 127;
  float v = src[(((size_t)s * B_ + bb) << 7) | dd];
  X[i]  = v;
  Xh[i] = (f16)v;
}

// one wave per row of (nrows, 128): row L2 norms
__global__ __launch_bounds__(256) void k_norms(const float* __restrict__ X,
                                               float* __restrict__ nrm, int nrows) {
  int wv = threadIdx.x >> 5, lane = threadIdx.x & 31;
  int row = blockIdx.x * 8 + wv;
  if (row >= nrows) return;
  float4 xv = ((const float4*)(X + (size_t)row * 128))[lane];
  float q = xv.x * xv.x + xv.y * xv.y + xv.z * xv.z + xv.w * xv.w;
#pragma unroll
  for (int m = 1; m <= 16; m <<= 1) q += __shfl_xor(q, m, 32);
  if (lane == 0) nrm[row] = sqrtf(q);
}

// out = LayerNorm(x + delta) * g + b ; writes f32 and f16
__global__ __launch_bounds__(256) void k_resid_ln(const float* __restrict__ X,
                                                  const float* __restrict__ Dl,
                                                  const float* __restrict__ g,
                                                  const float* __restrict__ bta,
                                                  float* __restrict__ outf,
                                                  f16* __restrict__ outh, int nrows) {
  int wv = threadIdx.x >> 5, lane = threadIdx.x & 31;
  int row = blockIdx.x * 8 + wv;
  if (row >= nrows) return;
  float4 xv = ((const float4*)(X  + (size_t)row * 128))[lane];
  float4 dv = ((const float4*)(Dl + (size_t)row * 128))[lane];
  float v0 = xv.x + dv.x, v1 = xv.y + dv.y, v2 = xv.z + dv.z, v3 = xv.w + dv.w;
  float s = v0 + v1 + v2 + v3;
  float q = v0 * v0 + v1 * v1 + v2 * v2 + v3 * v3;
#pragma unroll
  for (int m = 1; m <= 16; m <<= 1) { s += __shfl_xor(s, m, 32); q += __shfl_xor(q, m, 32); }
  float mu = s * (1.f / 128.f);
  float var = q * (1.f / 128.f) - mu * mu;
  float rs = rsqrtf(var + 1e-5f);
  int dd = lane << 2;
  float y0 = (v0 - mu) * rs * g[dd + 0] + bta[dd + 0];
  float y1 = (v1 - mu) * rs * g[dd + 1] + bta[dd + 1];
  float y2 = (v2 - mu) * rs * g[dd + 2] + bta[dd + 2];
  float y3 = (v3 - mu) * rs * g[dd + 3] + bta[dd + 3];
  ((float4*)(outf + (size_t)row * 128))[lane] = make_float4(y0, y1, y2, y3);
  v4h hv = { (f16)y0, (f16)y1, (f16)y2, (f16)y3 };
  ((v4h*)(outh + (size_t)row * 128))[lane] = hv;
}

// out = normalize(sigmoid(h) * x)
__global__ __launch_bounds__(256) void k_mask_norm(const float* __restrict__ Hc,
                                                   const float* __restrict__ X,
                                                   float* __restrict__ outf,
                                                   f16* __restrict__ outh, int nrows) {
  int wv = threadIdx.x >> 5, lane = threadIdx.x & 31;
  int row = blockIdx.x * 8 + wv;
  if (row >= nrows) return;
  float4 hv = ((const float4*)(Hc + (size_t)row * 128))[lane];
  float4 xv = ((const float4*)(X  + (size_t)row * 128))[lane];
  float v0 = xv.x / (1.f + __expf(-hv.x));
  float v1 = xv.y / (1.f + __expf(-hv.y));
  float v2 = xv.z / (1.f + __expf(-hv.z));
  float v3 = xv.w / (1.f + __expf(-hv.w));
  float q = v0 * v0 + v1 * v1 + v2 * v2 + v3 * v3;
#pragma unroll
  for (int m = 1; m <= 16; m <<= 1) q += __shfl_xor(q, m, 32);
  float inv = 1.f / fmaxf(sqrtf(q), 1e-12f);
  v0 *= inv; v1 *= inv; v2 *= inv; v3 *= inv;
  ((float4*)(outf + (size_t)row * 128))[lane] = make_float4(v0, v1, v2, v3);
  v4h ov = { (f16)v0, (f16)v1, (f16)v2, (f16)v3 };
  ((v4h*)(outh + (size_t)row * 128))[lane] = ov;
}

// ---------------------------------------------------------------------------
// Register-blocked WMMA GEMM: out = act(A(f16, MxK) @ W(f16, NxK)^T + bias)
// One wave per 32x64 output strip: 2 A frags x 4 B frags -> 8 WMMAs per
// K-step (4x A-reuse, 2x B-reuse). 8 waves/block. M%32==0, N%64==0.
// Output/activation config is compile-time: branch-free epilogue.
// ---------------------------------------------------------------------------
template <bool OUT32, bool OUT16, bool RELU>
__global__ __launch_bounds__(256) void k_gemm(const f16* __restrict__ A,
                                              const f16* __restrict__ W,
                                              const float* __restrict__ bias,
                                              float* __restrict__ out32,
                                              f16* __restrict__ out16,
                                              int M, int N, int K) {
  int wv = threadIdx.x >> 5, lane = threadIdx.x & 31;
  int tile = blockIdx.x * 8 + wv;
  int tilesN = N >> 6;                       // 64-wide column groups
  int tm = tile / tilesN, tg = tile - tm * tilesN;
  if (tm >= (M >> 5)) return;
  const f16* Ap0 = A + (size_t)tm * 32 * K;
  const f16* Ap1 = Ap0 + (size_t)16 * K;
  const f16* Wp  = W + (size_t)tg * 64 * K;
  v8f c[2][4];
#pragma unroll
  for (int r = 0; r < 2; ++r)
#pragma unroll
    for (int g = 0; g < 4; ++g) c[r][g] = {};
  for (int k = 0; k < K; k += 32) {
    v16h a0 = frag_a(Ap0 + k, K);
    v16h a1 = frag_a(Ap1 + k, K);
#pragma unroll
    for (int g = 0; g < 4; ++g) {
      v16h b = frag_bT(Wp + (size_t)(g * 16) * K + k, K);
      c[0][g] = wmma32(a0, b, c[0][g]);
      c[1][g] = wmma32(a1, b, c[1][g]);
    }
  }
  int half = lane >> 4;
#pragma unroll
  for (int r = 0; r < 2; ++r) {
#pragma unroll
    for (int g = 0; g < 4; ++g) {
      int col = tg * 64 + g * 16 + (lane & 15);
      float bv = bias[col];
#pragma unroll
      for (int i = 0; i < 8; ++i) {
        float v = c[r][g][i] + bv;
        if (RELU) v = fmaxf(v, 0.f);
        size_t row = (size_t)tm * 32 + r * 16 + i + half * 8;
        if (OUT32) out32[row * N + col] = v;
        if (OUT16) out16[row * N + col] = (f16)v;
      }
    }
  }
}

// ---------------------------------------------------------------------------
// Flash attention: one wave per (batch, head, 16-query tile).
// qkv f16 (B,S,384) with q/k/v at offsets 0/128/256, head h at +h*16.
// ---------------------------------------------------------------------------
__global__ __launch_bounds__(256) void k_attn(const f16* __restrict__ qkv,
                                              f16* __restrict__ attn_out) {
  __shared__ f16 pbuf[8][16 * 32];
  int wv = threadIdx.x >> 5, lane = threadIdx.x & 31;
  int unit = blockIdx.x * 8 + wv;               // B*H*(S/16) units, grid sized exactly
  int b  = unit >> 8;                           // / (8*32)
  int h  = (unit >> 5) & 7;
  int qt = unit & 31;
  int half = lane >> 4, col = lane & 15;

  const f16* qp = qkv + ((size_t)(b * S_ + qt * 16)) * 384 + h * 16;
  v16h aq = frag_a_pad(qp, 384, 16);            // HD=16 zero-padded to K=32

  float m[8], l[8];
  v8f o = {};
#pragma unroll
  for (int i = 0; i < 8; ++i) { m[i] = -1e30f; l[i] = 0.f; }

  for (int j = 0; j < 16; ++j) {                // 32 keys per iteration
    const f16* kp = qkv + ((size_t)(b * S_ + j * 32)) * 384 + 128 + h * 16;
    v8f z = {};
    v16h bk0 = frag_bT_pad(kp,            384, 16);
    v16h bk1 = frag_bT_pad(kp + 16 * 384, 384, 16);
    v8f s0 = wmma32(aq, bk0, z);
    v8f s1 = wmma32(aq, bk1, z);
#pragma unroll
    for (int i = 0; i < 8; ++i) {
      float x0 = s0[i] * 0.25f, x1 = s1[i] * 0.25f;   // 1/sqrt(HD)
      float mx = fmaxf(x0, x1);
#pragma unroll
      for (int mk = 1; mk <= 8; mk <<= 1) mx = fmaxf(mx, __shfl_xor(mx, mk, 32));
      float mn = fmaxf(m[i], mx);
      float fsc = __expf(m[i] - mn);
      float p0 = __expf(x0 - mn), p1 = __expf(x1 - mn);
      float rs = p0 + p1;
#pragma unroll
      for (int mk = 1; mk <= 8; mk <<= 1) rs += __shfl_xor(rs, mk, 32);
      l[i] = l[i] * fsc + rs;
      o[i] *= fsc;
      m[i] = mn;
      int r = i + half * 8;
      pbuf[wv][r * 32 + col]      = (f16)p0;
      pbuf[wv][r * 32 + 16 + col] = (f16)p1;
    }
    __syncthreads();                            // uniform across all 8 waves
    v16h ap = frag_a(&pbuf[wv][0], 32);
    const f16* vp = qkv + ((size_t)(b * S_ + j * 32)) * 384 + 256 + h * 16;
    v16h bv = frag_b_dir(vp, 384);
    o = wmma32(ap, bv, o);
    __syncthreads();
  }
#pragma unroll
  for (int i = 0; i < 8; ++i) {
    int r = i + half * 8;
    int q = qt * 16 + r;
    attn_out[((size_t)(b * S_ + q)) * D_ + h * 16 + col] = (f16)(o[i] / l[i]);
  }
}

// ---------------------------------------------------------------------------
// Pairwise clamped cosine: per-batch Gram via WMMA, 16x64 strip per wave.
// MODE 0: out = max(sim, 1e-6)
// MODE 1: out = clip(0.5*(max(sim,1e-6) + cs), 0.1, 0.9)
// ---------------------------------------------------------------------------
template <int MODE>
__global__ __launch_bounds__(256) void k_cos(const f16* __restrict__ Xh,
                                             const float* __restrict__ nrm,
                                             const float* __restrict__ cs_in,
                                             float* __restrict__ out) {
  int wv = threadIdx.x >> 5, lane = threadIdx.x & 31;
  int w = blockIdx.x * 8 + wv;                  // B*32*8 strips, grid sized exactly
  int b   = w >> 8;                             // / (32*8)
  int ti  = (w >> 3) & 31;
  int tjg = w & 7;                              // 64-wide column group
  const f16* Ap = Xh + ((size_t)(b * S_ + ti * 16)) * D_;
  const f16* Wp = Xh + ((size_t)(b * S_ + tjg * 64)) * D_;
  v8f c0 = {}, c1 = {}, c2 = {}, c3 = {};
#pragma unroll
  for (int k = 0; k < D_; k += 32) {
    v16h a = frag_a(Ap + k, D_);
    c0 = wmma32(a, frag_bT(Wp + (size_t)0  * D_ + k, D_), c0);
    c1 = wmma32(a, frag_bT(Wp + (size_t)16 * D_ + k, D_), c1);
    c2 = wmma32(a, frag_bT(Wp + (size_t)32 * D_ + k, D_), c2);
    c3 = wmma32(a, frag_bT(Wp + (size_t)48 * D_ + k, D_), c3);
  }
  int half = lane >> 4, col = lane & 15;
  v8f acc[4] = { c0, c1, c2, c3 };
#pragma unroll
  for (int g = 0; g < 4; ++g) {
    int jcol = tjg * 64 + g * 16 + col;
    float nj = nrm[b * S_ + jcol];
#pragma unroll
    for (int i = 0; i < 8; ++i) {
      int r = i + half * 8;
      float ni = nrm[b * S_ + ti * 16 + r];
      float sim = fmaxf(acc[g][i] / (ni * nj), 1e-6f);
      size_t idx = ((size_t)(b * S_ + ti * 16 + r)) * S_ + jcol;
      if (MODE) {
        out[idx] = fminf(fmaxf(0.5f * (sim + cs_in[idx]), 0.1f), 0.9f);
      } else {
        out[idx] = sim;
      }
    }
  }
}

// ---------------------------------------------------------------------------
extern "C" void kernel_launch(void* const* d_in, const int* in_sizes, int n_in,
                              void* d_out, int out_size, void* d_ws, size_t ws_size,
                              hipStream_t stream) {
  (void)in_sizes; (void)n_in; (void)out_size; (void)ws_size;
  const float* src  = (const float*)d_in[0];
  const float* Wqkv = (const float*)d_in[1];
  const float* bqkv = (const float*)d_in[2];
  const float* Wo   = (const float*)d_in[3];
  const float* bo   = (const float*)d_in[4];
  const float* ln1g = (const float*)d_in[5];
  const float* ln1b = (const float*)d_in[6];
  const float* W1   = (const float*)d_in[7];
  const float* b1   = (const float*)d_in[8];
  const float* W2   = (const float*)d_in[9];
  const float* b2   = (const float*)d_in[10];
  const float* ln2g = (const float*)d_in[11];
  const float* ln2b = (const float*)d_in[12];
  float* out = (float*)d_out;

  const size_t MR = (size_t)B_ * S_;            // 32768 rows

  char* ws = (char*)d_ws;
  size_t off = 0;
  auto alloc = [&](size_t bytes) -> void* {
    void* p = ws + off;
    off += (bytes + 255) & ~(size_t)255;
    return p;
  };
  float* X     = (float*)alloc(MR * D_ * 4);
  f16*   Xh    = (f16*)  alloc(MR * D_ * 2);
  float* CS    = (float*)alloc((size_t)B_ * S_ * S_ * 4);
  float* Hf    = (float*)alloc(MR * D_ * 4);
  f16*   Hh    = (f16*)  alloc(MR * D_ * 2);
  f16*   QKVh  = (f16*)  alloc(MR * 3 * D_ * 2);
  f16*   ATTNh = (f16*)  alloc(MR * D_ * 2);
  float* TMP   = (float*)alloc(MR * D_ * 4);
  f16*   FFh   = (f16*)  alloc(MR * FF_ * 2);
  f16*   Wqkvh = (f16*)  alloc((size_t)L_ * 3 * D_ * D_ * 2);
  f16*   Woh   = (f16*)  alloc((size_t)L_ * D_ * D_ * 2);
  f16*   W1h   = (f16*)  alloc((size_t)L_ * FF_ * D_ * 2);
  f16*   W2h   = (f16*)  alloc((size_t)L_ * D_ * FF_ * 2);
  float* NORMS = (float*)alloc(MR * 4);
  float* OUTN  = TMP;    // reuse: free after last layer
  f16*   OUTNh = ATTNh;  // reuse

  // ---- weight conversion f32 -> f16 ----
  {
    int n;
    n = L_ * 3 * D_ * D_; k_cvt<<<(n + 255) / 256, 256, 0, stream>>>(Wqkv, Wqkvh, n);
    n = L_ * D_ * D_;     k_cvt<<<(n + 255) / 256, 256, 0, stream>>>(Wo,   Woh,   n);
    n = L_ * FF_ * D_;    k_cvt<<<(n + 255) / 256, 256, 0, stream>>>(W1,   W1h,   n);
    n = L_ * D_ * FF_;    k_cvt<<<(n + 255) / 256, 256, 0, stream>>>(W2,   W2h,   n);
  }

  // ---- x = transpose(src); cs = pairwise_cos(x) ----
  k_transpose<<<(int)(MR * D_ / 256), 256, 0, stream>>>(src, X, Xh);
  k_norms<<<(int)(MR / 8), 256, 0, stream>>>(X, NORMS, (int)MR);
  k_cos<0><<<B_ * 32 * 8 / 8, 256, 0, stream>>>(Xh, NORMS, nullptr, CS);

  // h = x
  hipMemcpyAsync(Hf, X,  MR * D_ * 4, hipMemcpyDeviceToDevice, stream);
  hipMemcpyAsync(Hh, Xh, MR * D_ * 2, hipMemcpyDeviceToDevice, stream);

  // ---- encoder layers ----
  const int M = (int)MR;
  for (int l = 0; l < L_; ++l) {
    // qkv = h @ Wqkv^T + bqkv  (N=384 -> 6 col groups)
    k_gemm<false, true, false><<<(M / 32) * (3 * D_ / 64) / 8, 256, 0, stream>>>(
        Hh, Wqkvh + (size_t)l * 3 * D_ * D_, bqkv + l * 3 * D_,
        nullptr, QKVh, M, 3 * D_, D_);
    // attention
    k_attn<<<B_ * H_ * (S_ / 16) / 8, 256, 0, stream>>>(QKVh, ATTNh);
    // o = attn @ Wo^T + bo  (N=128 -> 2 col groups)
    k_gemm<true, false, false><<<(M / 32) * (D_ / 64) / 8, 256, 0, stream>>>(
        ATTNh, Woh + (size_t)l * D_ * D_, bo + l * D_,
        TMP, nullptr, M, D_, D_);
    // h = LN1(h + o)
    k_resid_ln<<<M / 8, 256, 0, stream>>>(Hf, TMP, ln1g + l * D_, ln1b + l * D_,
                                          Hf, Hh, M);
    // ff = relu(h @ W1^T + b1)  (N=1024 -> 16 col groups)
    k_gemm<false, true, true><<<(M / 32) * (FF_ / 64) / 8, 256, 0, stream>>>(
        Hh, W1h + (size_t)l * FF_ * D_, b1 + l * FF_,
        nullptr, FFh, M, FF_, D_);
    // f2 = ff @ W2^T + b2  (K=1024)
    k_gemm<true, false, false><<<(M / 32) * (D_ / 64) / 8, 256, 0, stream>>>(
        FFh, W2h + (size_t)l * D_ * FF_, b2 + l * D_,
        TMP, nullptr, M, D_, FF_);
    // h = LN2(h + f2)
    k_resid_ln<<<M / 8, 256, 0, stream>>>(Hf, TMP, ln2g + l * D_, ln2b + l * D_,
                                          Hf, Hh, M);
  }

  // ---- out = normalize(sigmoid(h) * x); final = clip(0.5*(cos(out)+cs)) ----
  k_mask_norm<<<M / 8, 256, 0, stream>>>(Hf, X, OUTN, OUTNh, M);
  k_norms<<<M / 8, 256, 0, stream>>>(OUTN, NORMS, M);
  k_cos<1><<<B_ * 32 * 8 / 8, 256, 0, stream>>>(OUTNh, NORMS, CS, out);
}
